// SimpleNetwork_25821343384203
// MI455X (gfx1250) — compile-verified, matrix operand order checked by reading, not run on hardware
//
#include <hip/hip_runtime.h>
#include <math.h>

// ---------------- problem constants (from the reference) ----------------
#define N_NODES   50000
#define N_EDGES   1600000
#define IN_DIM    16
#define MUL       16
#define NB        10
#define NUM_GRAPHS 16

#define INV_SQRT_NB   0.31622776601f   // 1/sqrt(10)
#define SQRT_NB       3.16227766017f
#define WSCALE        0.017677669529f  // 1/sqrt(100) * 1/sqrt(32)
#define SQRT3         1.73205080757f
#define INV_SQRT3     0.57735026919f

typedef _Float16 v16h __attribute__((ext_vector_type(16)));
typedef _Float16 h8v  __attribute__((ext_vector_type(8)));
typedef _Float16 h2v  __attribute__((ext_vector_type(2)));
typedef float    v8f  __attribute__((ext_vector_type(8)));

__device__ __forceinline__ v8f wmma_f16(v16h a, v16h b, v8f c) {
    // D = A(16x32 f16) * B(32x16 f16) + C(16x16 f32)
    return __builtin_amdgcn_wmma_f32_16x16x32_f16(false, a, false, b,
                                                  (short)0, c, false, false);
}

__device__ __forceinline__ v16h frag_join(h8v lo, h8v hh) {
    return __builtin_shufflevector(lo, hh, 0, 1, 2, 3, 4, 5, 6, 7,
                                           8, 9, 10, 11, 12, 13, 14, 15);
}

__device__ __forceinline__ h8v h8_zero() {
    h8v z;
#pragma unroll
    for (int j = 0; j < 8; ++j) z[j] = (_Float16)0.0f;
    return z;
}

// fragment from LDS row with 128-half K-stride:
// halves 0..7 -> K = 32t + 8*hi + j ; halves 8..15 -> K = 32t + 16 + 8*hi + j
__device__ __forceinline__ v16h load_frag(const _Float16* row, int t, int hi) {
    h8v lo = *(const h8v*)(row + t * 32 + hi * 8);
    h8v hh = *(const h8v*)(row + t * 32 + 16 + hi * 8);
    return frag_join(lo, hh);
}

__device__ __forceinline__ void wave_lds_fence() {
    asm volatile("s_wait_dscnt 0" ::: "memory");   // CDNA5 split DS counter
}

__device__ __forceinline__ float fast_silu(float v) {
    return v * __builtin_amdgcn_rcpf(1.0f + __expf(-v));
}

// ---------------- zero-fill ----------------
__global__ void e3nn_zero_kernel(float* p, int n) {
    for (int i = blockIdx.x * blockDim.x + threadIdx.x; i < n;
         i += gridDim.x * blockDim.x)
        p[i] = 0.0f;
}

// ---------------- fused edge kernel (LAYER = 1 or 2) ----------------
// All GEMMs are computed TRANSPOSED (D^T = B^T @ A^T) so the C-layout gives
// each lane 8 K-contiguous outputs for a single edge -> b128 LDS stores and
// register-resident per-edge metadata for the scatter phase.
template <int LAYER>
__global__ __launch_bounds__(256)
void e3nn_edge_kernel(const float* __restrict__ x,
                      const float* __restrict__ edge_vec,
                      const float* __restrict__ w1,
                      const float* __restrict__ w2sel,    // w2 (L1) or w2b (L2)
                      const float* __restrict__ wmix0,
                      const float* __restrict__ wmix1,
                      const int*   __restrict__ edge_src,
                      const int*   __restrict__ edge_dst,
                      float* __restrict__ n0,
                      float* __restrict__ n1,
                      float* __restrict__ nf) {
    __shared__ __align__(16) _Float16 s_w1t [112 * 16];   // [hidden][k(pad16)]
    __shared__ __align__(16) _Float16 s_wt  [32 * 128];   // [col][k(pad128)]
    __shared__ __align__(16) _Float16 s_mx0 [MUL * 16];   // [m][k]
    __shared__ __align__(16) _Float16 s_mx1 [MUL * 16];
    __shared__ __align__(16) _Float16 s_h   [8][16 * 128];// per-wave hT tile
    __shared__ __align__(16) _Float16 s_wAB [8][16 * 32]; // per-wave TP weights

    const int tid = threadIdx.x;

    // ---- stage weights into LDS (once per block) ----
    for (int i = tid; i < 112 * 16; i += 256) {
        int col = i >> 4, k = i & 15;
        float v = (k < NB && col < 100) ? w1[k * 100 + col] : 0.0f;
        s_w1t[i] = (_Float16)v;
    }
    for (int i = tid; i < 32 * 128; i += 256) {
        int col = i >> 7, k = i & 127;
        float v = (k < 100) ? w2sel[k * 32 + col] : 0.0f;
        s_wt[i] = (_Float16)v;
    }
    if (LAYER == 1) {
        for (int i = tid; i < MUL * 16; i += 256) {
            int m = i >> 4, k = i & 15;
            s_mx0[m * 16 + k] = (_Float16)wmix0[k * MUL + m];
            s_mx1[m * 16 + k] = (_Float16)wmix1[k * MUL + m];
        }
    }
    __syncthreads();

    const int lane = tid & 31;
    const int wave = tid >> 5;
    const int Mrow = lane & 15;     // A-row / B-col index; also this lane's edge
    const int hi   = lane >> 4;

    // ---- hoisted loop-invariant A fragments (transposed weights) ----
    v16h aw1[7];                                     // w1^T tiles (K=10 pad 32)
#pragma unroll
    for (int n = 0; n < 7; ++n) {
        h8v lo = *(const h8v*)&s_w1t[(16 * n + Mrow) * 16 + hi * 8];
        aw1[n] = frag_join(lo, h8_zero());
    }
    v16h aw2[8];                                     // w2^T tiles (K=100 pad 128)
#pragma unroll
    for (int t = 0; t < 4; ++t)
#pragma unroll
        for (int n = 0; n < 2; ++n)
            aw2[t * 2 + n] = load_frag(&s_wt[(16 * n + Mrow) * 128], t, hi);
    v16h am0, am1;                                   // wmix^T (K=16 pad 32)
    if (LAYER == 1) {
        h8v l0 = *(const h8v*)&s_mx0[Mrow * 16 + hi * 8];
        h8v l1 = *(const h8v*)&s_mx1[Mrow * 16 + hi * 8];
        am0 = frag_join(l0, h8_zero());
        am1 = frag_join(l1, h8_zero());
    }

    const int numGroups = (N_EDGES + 127) / 128;     // 128 edges per block-iter
    for (int g = blockIdx.x; g < numGroups; g += gridDim.x) {
        const int base = g * 128;

        // ---- per-lane edge setup (lane owns edge Mrow of its wave tile) ----
        const int eReal = base + wave * 16 + Mrow;
        const int eA = (eReal < N_EDGES) ? eReal : (N_EDGES - 1);
        const int srcL = edge_src[eA];
        const int dstL = edge_dst[eA];
        float vx = edge_vec[eA * 3 + 0];
        float vy = edge_vec[eA * 3 + 1];
        float vz = edge_vec[eA * 3 + 2];
        float r  = sqrtf(vx * vx + vy * vy + vz * vz);
        float isc = SQRT3 * __builtin_amdgcn_rcpf(r + 1e-9f);
        const float sh0 = vx * isc, sh1 = vy * isc, sh2 = vz * isc;

        // cosine soft-one-hot basis, scaled by sqrt(NB)
        float embv[NB];
        {
            const float stepv = 3.5f / 11.0f;
            const float istep = 11.0f / 3.5f;
#pragma unroll
            for (int j = 0; j < NB; ++j) {
                float diff = (r - (float)(j + 1) * stepv) * istep;
                bool in = (diff > -1.0f) && (diff < 1.0f);
                embv[j] = in ? __cosf(1.57079632679f * diff) * SQRT_NB : 0.0f;
            }
        }

        // B fragment: emb^T [K(pad 10->32) x 16 edges]
        v16h bemb;
#pragma unroll
        for (int j = 0; j < 16; ++j) bemb[j] = (_Float16)0.0f;
        if (hi == 0) {
#pragma unroll
            for (int j = 0; j < 8; ++j) bemb[j] = (_Float16)embv[j];
        } else {
            bemb[0] = (_Float16)embv[8];
            bemb[1] = (_Float16)embv[9];
        }

        // GEMM1 (transposed): h^T[112x16] = w1^T @ emb^T   (7 WMMAs)
        v8f acc[7];
#pragma unroll
        for (int n = 0; n < 7; ++n)
#pragma unroll
            for (int q = 0; q < 8; ++q) acc[n][q] = 0.0f;
#pragma unroll
        for (int n = 0; n < 7; ++n)
            acc[n] = wmma_f16(aw1[n], bemb, acc[n]);

        // silu -> LDS [edge][K] tile; lane holds 8 contiguous K per acc
#pragma unroll
        for (int n = 0; n < 7; ++n) {
            h8v hv;
#pragma unroll
            for (int q = 0; q < 8; ++q)
                hv[q] = (_Float16)fast_silu(acc[n][q] * INV_SQRT_NB);
            *(h8v*)&s_h[wave][Mrow * 128 + 16 * n + 8 * hi] = hv;
        }
        *(h8v*)&s_h[wave][Mrow * 128 + 112 + 8 * hi] = h8_zero();  // K pad
        wave_lds_fence();

        // GEMM2 (transposed): wX^T[32x16] = w2^T @ h^T   (8 WMMAs)
        v8f accW[2];
#pragma unroll
        for (int n = 0; n < 2; ++n)
#pragma unroll
            for (int q = 0; q < 8; ++q) accW[n][q] = 0.0f;
        const _Float16* hrow = &s_h[wave][Mrow * 128];
#pragma unroll
        for (int t = 0; t < 4; ++t) {
            v16h bh = load_frag(hrow, t, hi);
#pragma unroll
            for (int n = 0; n < 2; ++n)
                accW[n] = wmma_f16(aw2[t * 2 + n], bh, accW[n]);
        }
#pragma unroll
        for (int n = 0; n < 2; ++n) {
            h8v wv;
#pragma unroll
            for (int q = 0; q < 8; ++q)
                wv[q] = (_Float16)(accW[n][q] * WSCALE);   // folds 0.1/sqrt(32)
            *(h8v*)&s_wAB[wave][Mrow * 32 + 16 * n + 8 * hi] = wv;
        }

        // x-mix GEMMs (layer 1): xm^T = wmix^T @ x[src]^T  (2 WMMAs)
        v8f xm0a, xm1a;
        if (LAYER == 1) {
            const float4* xp = (const float4*)(x + srcL * IN_DIM + hi * 8);
            float4 c0 = xp[0], c1 = xp[1];
            h8v xr;
            xr[0] = (_Float16)c0.x; xr[1] = (_Float16)c0.y;
            xr[2] = (_Float16)c0.z; xr[3] = (_Float16)c0.w;
            xr[4] = (_Float16)c1.x; xr[5] = (_Float16)c1.y;
            xr[6] = (_Float16)c1.z; xr[7] = (_Float16)c1.w;
            v16h bx = frag_join(xr, h8_zero());      // K = 16..31 zero
            v8f z;
#pragma unroll
            for (int q = 0; q < 8; ++q) z[q] = 0.0f;
            xm0a = wmma_f16(am0, bx, z);
            xm1a = wmma_f16(am1, bx, z);
        }
        wave_lds_fence();

        // ---- scatter: lane owns (edge = Mrow, m = 8*hi + q) ----
        if (eReal < N_EDGES) {
            if (LAYER == 1) {
#pragma unroll
                for (int q = 0; q < 8; ++q) {
                    const int m = 8 * hi + q;
                    const h2v wp =
                        *(const h2v*)&s_wAB[wave][Mrow * 32 + 2 * m];
                    float m0v = xm0a[q] * (float)wp[0];
                    atomicAdd(&n0[dstL * MUL + m], m0v);
                    float c = xm1a[q] * (float)wp[1];
                    atomicAdd(&n1[(dstL * MUL + m) * 3 + 0], c * sh0);
                    atomicAdd(&n1[(dstL * MUL + m) * 3 + 1], c * sh1);
                    atomicAdd(&n1[(dstL * MUL + m) * 3 + 2], c * sh2);
                }
            } else {
                // contiguous vector gathers of n0/n1 for this lane's m-range
                const float4* p0 = (const float4*)(n0 + srcL * MUL + 8 * hi);
                float4 a0 = p0[0], a1 = p0[1];
                float n0v[8] = { a0.x, a0.y, a0.z, a0.w,
                                 a1.x, a1.y, a1.z, a1.w };
                const float4* p1 =
                    (const float4*)(n1 + (srcL * MUL + 8 * hi) * 3);
                float n1v[24];
#pragma unroll
                for (int j = 0; j < 6; ++j) {
                    float4 t4 = p1[j];
                    n1v[4 * j + 0] = t4.x; n1v[4 * j + 1] = t4.y;
                    n1v[4 * j + 2] = t4.z; n1v[4 * j + 3] = t4.w;
                }
#pragma unroll
                for (int q = 0; q < 8; ++q) {
                    const int m = 8 * hi + q;
                    const h2v wp =
                        *(const h2v*)&s_wAB[wave][Mrow * 32 + 2 * m];
                    float v0 = n0v[q] * (float)wp[0];
                    float d  = n1v[3 * q + 0] * sh0 +
                               n1v[3 * q + 1] * sh1 +
                               n1v[3 * q + 2] * sh2;
                    float v1 = d * (float)wp[1] * INV_SQRT3;
                    atomicAdd(&nf[dstL * MUL + m], v0 + v1);
                }
            }
        }
    }
}

// ---------------- node readout + graph pooling ----------------
__global__ void e3nn_readout_kernel(const float* __restrict__ nf,
                                    const float* __restrict__ w_out,
                                    const float* __restrict__ factor,
                                    const int*   __restrict__ batch,
                                    float* __restrict__ out) {
    int n = blockIdx.x * blockDim.x + threadIdx.x;
    if (n >= N_NODES) return;
    float a = 0.0f;
#pragma unroll
    for (int m = 0; m < MUL; ++m) a = fmaf(nf[n * MUL + m], w_out[m], a);
    atomicAdd(&out[batch[n]], a * factor[0]);
}

// ---------------- launch ----------------
extern "C" void kernel_launch(void* const* d_in, const int* in_sizes, int n_in,
                              void* d_out, int out_size, void* d_ws, size_t ws_size,
                              hipStream_t stream) {
    const float* x        = (const float*)d_in[0];
    const float* edge_vec = (const float*)d_in[1];
    const float* w1       = (const float*)d_in[2];
    const float* w2       = (const float*)d_in[3];
    const float* w2b      = (const float*)d_in[4];
    const float* wmix0    = (const float*)d_in[5];
    const float* wmix1    = (const float*)d_in[6];
    const float* w_out    = (const float*)d_in[7];
    const float* factor   = (const float*)d_in[8];
    const int*   edge_src = (const int*)d_in[9];
    const int*   edge_dst = (const int*)d_in[10];
    const int*   batch    = (const int*)d_in[11];
    float* out = (float*)d_out;

    float* n0 = (float*)d_ws;                       // [N_NODES*16]
    float* n1 = n0 + (size_t)N_NODES * MUL;         // [N_NODES*16*3]
    float* nf = n1 + (size_t)N_NODES * MUL * 3;     // [N_NODES*16]
    const int ws_elems = N_NODES * MUL * 5;         // 4,000,000 floats

    e3nn_zero_kernel<<<2048, 256, 0, stream>>>(n0, ws_elems);
    e3nn_zero_kernel<<<1, 32, 0, stream>>>(out, NUM_GRAPHS);

    e3nn_edge_kernel<1><<<2048, 256, 0, stream>>>(
        x, edge_vec, w1, w2, wmix0, wmix1, edge_src, edge_dst, n0, n1, nf);
    e3nn_edge_kernel<2><<<2048, 256, 0, stream>>>(
        x, edge_vec, w1, w2b, wmix0, wmix1, edge_src, edge_dst, n0, n1, nf);

    e3nn_readout_kernel<<<(N_NODES + 255) / 256, 256, 0, stream>>>(
        nf, w_out, factor, batch, out);
}
